// MultiHeadAttentionLinear_79499844649675
// MI455X (gfx1250) — matrix operand order repaired and checked by reading, not compiled, not run on hardware
//
#include <hip/hip_runtime.h>

// ---------------------------------------------------------------------------
// MultiHeadAttentionLinear for MI455X (gfx1250, wave32, WMMA bf16 + TDM)
//   B=8, Lq=Lk=1024, D=512, H=8, d=512
// ---------------------------------------------------------------------------

typedef __bf16 bf16_t;
typedef __attribute__((ext_vector_type(16))) __bf16 v16bf;
typedef __attribute__((ext_vector_type(8)))  __bf16 v8bf;
typedef __attribute__((ext_vector_type(8)))  float  v8f;
typedef unsigned int u32x4 __attribute__((ext_vector_type(4)));
typedef int          i32x8 __attribute__((ext_vector_type(8)));
typedef int          i32x4 __attribute__((ext_vector_type(4)));

#define SHUF16(a, b) __builtin_shufflevector(a, b, 0,1,2,3,4,5,6,7,8,9,10,11,12,13,14,15)

static __device__ __forceinline__ v8f wmma_bf16(v16bf a, v16bf b, v8f c) {
    return __builtin_amdgcn_wmma_f32_16x16x32_bf16(
        /*neg_a=*/false, a, /*neg_b=*/false, b,
        /*c_mod=*/(short)0, c, /*reuse_a=*/false, /*reuse_b=*/false);
}

// TDM: 2D tile load Global -> LDS. D# per CDNA5 ISA §8 (data_size = 2 bytes).
// This toolchain exposes the 6-arg builtin:
//   (uint32x4 g0, int32x8 g1, int32x4 g2, int32x4 g3, int32x8 g4, i32 cpol)
static __device__ __forceinline__ void tdm_load_tile_2d(
    unsigned lds_addr, const void* gptr,
    unsigned tensor_d0, unsigned tensor_d1,
    unsigned tile_d0, unsigned tile_d1, unsigned stride0)
{
    const unsigned long long ga = (unsigned long long)(uintptr_t)gptr;
    u32x4 g0;
    g0[0] = 1u;                                         // count=1, no gather
    g0[1] = lds_addr;                                   // LDS byte address
    g0[2] = (unsigned)ga;                               // global_addr[31:0]
    g0[3] = (unsigned)((ga >> 32) & 0x01FFFFFFu) | (2u << 30);  // type=2
    i32x8 g1;
    g1[0] = 0x00010000;                                 // data_size=2B, mask=0
    g1[1] = (int)(tensor_d0 << 16);                     // tensor_dim0[15:0]
    g1[2] = (int)((tensor_d0 >> 16) | (tensor_d1 << 16));
    g1[3] = (int)((tensor_d1 >> 16) | (tile_d0 << 16)); // tile_dim0
    g1[4] = (int)(tile_d1 & 0xFFFF);                    // tile_dim1 (tile_dim2=0)
    g1[5] = (int)stride0;                               // tensor_dim0_stride
    g1[6] = 0;
    g1[7] = 0;
    i32x4 gz4 = {0, 0, 0, 0};
    i32x8 gz8 = {0, 0, 0, 0, 0, 0, 0, 0};
    __builtin_amdgcn_tensor_load_to_lds(g0, g1, gz4, gz4, gz8, 0);
}

// -------------------------- constants --------------------------------------
#define BB   8
#define HH   8
#define LL   1024   // Lq == Lk
#define DD   512    // model dim == per-head dim
#define QT   32     // query rows per block
#define KT   32     // key block
#define NBK  (LL / KT)

// ---------------------------------------------------------------------------
// Prep: fp32 -> bf16 conversion (optionally fused add)
// ---------------------------------------------------------------------------
__global__ void cvt_add_kernel(const float* __restrict__ a,
                               const float* __restrict__ b,
                               bf16_t* __restrict__ o, int n) {
    int i = blockIdx.x * blockDim.x + threadIdx.x;
    int stride = gridDim.x * blockDim.x;
    for (; i < n; i += stride) o[i] = (bf16_t)(a[i] + b[i]);
}

__global__ void cvt_kernel(const float* __restrict__ a,
                           bf16_t* __restrict__ o, int n) {
    int i = blockIdx.x * blockDim.x + threadIdx.x;
    int stride = gridDim.x * blockDim.x;
    for (; i < n; i += stride) o[i] = (bf16_t)a[i];
}

// ---------------------------------------------------------------------------
// Projection GEMM:  C[m][n] = sum_k A[m][k] * W[n][k] + bias[n]
//   M = B*L = 8192, K = 512, N = 4096 (QKV) or 512 (XP)
// 256 threads = 8 waves; each wave computes a 16x64 tile; block = 16x512.
// mode 0: Q -> bf16 [H,B,L,d]      mode 1: K -> bf16 [H,B,L,d]
// mode 2: V -> bf16 [H,B,d,L] (transposed)   mode 3: XP -> f32 [B,L,d]
// ---------------------------------------------------------------------------
__global__ __launch_bounds__(256) void proj_kernel(
    const bf16_t* __restrict__ A, const bf16_t* __restrict__ W,
    const float* __restrict__ bias, void* __restrict__ outv,
    int N, int mode)
{
    const int tid  = threadIdx.x;
    const int lane = tid & 31;
    const int w    = tid >> 5;
    const int nb   = N >> 9;                   // N / 512
    const int bm   = blockIdx.x / nb;
    const int bn   = blockIdx.x % nb;
    const int mbase   = bm << 4;               // 16 rows
    const int colbase = (bn << 9) + w * 64;    // 64 cols per wave
    const int rlo = lane & 15;
    const int lhi = lane >> 4;

    v8f acc[4] = {};

#pragma unroll 4
    for (int kk = 0; kk < DD; kk += 32) {
        const bf16_t* arow = A + (size_t)(mbase + rlo) * DD + kk + lhi * 16;
        v16bf af = *(const v16bf*)(arow);
#pragma unroll
        for (int t = 0; t < 4; ++t) {
            const bf16_t* wrow =
                W + (size_t)(colbase + t * 16 + rlo) * DD + kk + lhi * 16;
            v16bf bfrag = *(const v16bf*)wrow;
            acc[t] = wmma_bf16(af, bfrag, acc[t]);
        }
    }

#pragma unroll
    for (int t = 0; t < 4; ++t) {
        const int n  = colbase + t * 16 + rlo;
        const float bb = bias ? bias[n] : 0.0f;
#pragma unroll
        for (int j = 0; j < 8; ++j) {
            const int m    = mbase + j + lhi * 8;
            const float v  = acc[t][j] + bb;
            const int bidx = m >> 10;          // batch
            const int l    = m & (LL - 1);     // position
            if (mode <= 1) {                   // Q or K: [H,B,L,d] bf16
                const int hh = n >> 9, dc = n & (DD - 1);
                bf16_t* o = (bf16_t*)outv;
                o[(((size_t)hh * BB + bidx) * LL + l) * DD + dc] = (bf16_t)v;
            } else if (mode == 2) {            // V transposed: [H,B,d,L] bf16
                const int hh = n >> 9, dc = n & (DD - 1);
                bf16_t* o = (bf16_t*)outv;
                o[(((size_t)hh * BB + bidx) * DD + dc) * LL + l] = (bf16_t)v;
            } else {                           // XP: [B,L,d] f32
                float* o = (float*)outv;
                o[((size_t)bidx * LL + l) * DD + n] = v;
            }
        }
    }
}

// ---------------------------------------------------------------------------
// Flash attention + gated residual, TDM double-buffered K/V staging in LDS.
// One block per (h, b, 32-query tile). 256 threads = 8 waves:
//   wave w -> q-subtile sq = w>>2 (16 rows), d-slice ds = w&3 (128 cols).
// Per 32-key block: wave0 TDM-prefetches next K/V tiles; partial S = Q K^T
// from LDS K (WMMA), ds_add_f32 reduce across d-slices; online softmax per
// subtile; P*V WMMA from LDS V into 16x128 f32 accumulators.
// out[b][h][lq][dc] = (g * softmax(QK^T)V + XP) / (g + 1)
// ---------------------------------------------------------------------------
__global__ __launch_bounds__(256) void attn_kernel(
    const bf16_t* __restrict__ Qbf, const bf16_t* __restrict__ Kbf,
    const bf16_t* __restrict__ Vt,  const float* __restrict__ XP,
    const int* __restrict__ y_mask, const float* __restrict__ gamma,
    float* __restrict__ out)
{
    __shared__ alignas(32) bf16_t Klds[2][KT][DD];    // 2 x 32KB
    __shared__ alignas(32) bf16_t Vlds[2][DD][KT];    // 2 x 32KB
    __shared__ alignas(16) float  S[2][16][KT];       // 4KB (per subtile)
    __shared__ alignas(32) bf16_t P[2][16][KT];       // 2KB
    __shared__ float mrow[2][16], lrow[2][16], frow[2][16];

    const int tid  = threadIdx.x;
    const int lane = tid & 31;
    const int w    = tid >> 5;
    const int sq   = w >> 2;          // q-subtile (0/1)
    const int dwave = (w & 3) * 128;  // this wave's 128-wide d-slice

    const int blk = blockIdx.x;
    const int h  = blk >> 8;          // / (B * 32)
    const int b  = (blk >> 5) & 7;
    const int qt = blk & 31;
    const int qbase = qt << 5;        // 32 rows

    const int hb = h * BB + b;
    const bf16_t* Qp = Qbf + (size_t)hb * LL * DD;
    const bf16_t* Kp = Kbf + (size_t)hb * LL * DD;
    const bf16_t* Vp = Vt  + (size_t)hb * DD * LL;

    const int rlo = lane & 15;
    const int lhi = lane >> 4;
    const int qrow0 = qbase + sq * 16;

    // Q fragments: 4 k-steps of 32 over this wave's 128-wide d-slice
    v16bf qfrag[4];
#pragma unroll
    for (int ks = 0; ks < 4; ++ks) {
        const int dbase = dwave + ks * 32;
        const bf16_t* qrow = Qp + (size_t)(qrow0 + rlo) * DD + dbase + lhi * 16;
        qfrag[ks] = *(const v16bf*)(qrow);
    }

    v8f acc[8] = {};                  // 16 x 128 f32 accumulator
    if (tid < 32) { mrow[tid >> 4][tid & 15] = -1e30f; lrow[tid >> 4][tid & 15] = 0.0f; }

    // Preload key-block 0 via TDM (wave 0 only; wave-uniform branch)
    if (w == 0) {
        tdm_load_tile_2d((unsigned)(uintptr_t)&Klds[0][0][0], Kp,
                         DD, KT, DD, KT, DD);
        tdm_load_tile_2d((unsigned)(uintptr_t)&Vlds[0][0][0], Vp,
                         KT, DD, KT, DD, LL);
    }

    for (int ib = 0; ib < NBK; ++ib) {
        const int kb  = ib * KT;
        const int cur = ib & 1;
        const int nxt = cur ^ 1;

        // ---- zero S (1024 floats) + prefetch next tiles + wait current -----
        {
            float* s1 = &S[0][0][0];
            s1[tid] = 0.0f; s1[tid + 256] = 0.0f;
            s1[tid + 512] = 0.0f; s1[tid + 768] = 0.0f;
        }
        if (w == 0) {
            if (ib + 1 < NBK) {
                const int kn = kb + KT;
                tdm_load_tile_2d((unsigned)(uintptr_t)&Klds[nxt][0][0],
                                 Kp + (size_t)kn * DD, DD, KT, DD, KT, DD);
                tdm_load_tile_2d((unsigned)(uintptr_t)&Vlds[nxt][0][0],
                                 Vp + kn, KT, DD, KT, DD, LL);
                __builtin_amdgcn_s_wait_tensorcnt(2);   // current pair done
            } else {
                __builtin_amdgcn_s_wait_tensorcnt(0);
            }
        }
        __syncthreads();   // #1: S zeroed, buf[cur] staged

        // ---- partial S = Q K^T over this wave's 128-wide d-slice -----------
        v8f c2[2] = {};
#pragma unroll
        for (int ks = 0; ks < 4; ++ks) {
            const int dbase = dwave + ks * 32;
#pragma unroll
            for (int nt = 0; nt < 2; ++nt) {
                const bf16_t* krow = &Klds[cur][nt * 16 + rlo][dbase + lhi * 16];
                v16bf bk = *(const v16bf*)krow;
                c2[nt] = wmma_bf16(qfrag[ks], bk, c2[nt]);
            }
        }
#pragma unroll
        for (int nt = 0; nt < 2; ++nt)
#pragma unroll
            for (int j = 0; j < 8; ++j)
                atomicAdd(&S[sq][j + lhi * 8][nt * 16 + rlo], c2[nt][j]);
        __syncthreads();   // #2: S complete

        // ---- online softmax (waves 0 and 4, lanes 0..15: one row each) -----
        if ((w & 3) == 0 && lane < 16) {
            const int row = lane;
            float sv[KT];
            float mb = -1e30f;
#pragma unroll
            for (int kk = 0; kk < KT; ++kk) {
                const float s = y_mask[b * LL + kb + kk] ? S[sq][row][kk] : -1e30f;
                sv[kk] = s;
                mb = fmaxf(mb, s);
            }
            const float mo = mrow[sq][row];
            const float mn = fmaxf(mo, mb);
            const float f  = __expf(mo - mn);
            float lsum = 0.0f;
#pragma unroll
            for (int kk = 0; kk < KT; ++kk) {
                const float p = __expf(sv[kk] - mn);
                lsum += p;
                P[sq][row][kk] = (bf16_t)p;
            }
            mrow[sq][row] = mn;
            lrow[sq][row] = lrow[sq][row] * f + lsum;
            frow[sq][row] = f;
        }
        __syncthreads();   // #3: P / factors ready

        // ---- rescale accumulators + P*V on this wave's d-slice -------------
        float fj[8];
#pragma unroll
        for (int j = 0; j < 8; ++j) fj[j] = frow[sq][j + lhi * 8];

        const bf16_t* prow = &P[sq][rlo][0];
        v16bf pfrag = *(const v16bf*)(prow + lhi * 16);

#pragma unroll
        for (int t = 0; t < 8; ++t) {
#pragma unroll
            for (int j = 0; j < 8; ++j) acc[t][j] *= fj[j];
            const bf16_t* vrow = &Vlds[cur][dwave + t * 16 + rlo][lhi * 16];
            v16bf bv = *(const v16bf*)vrow;
            acc[t] = wmma_bf16(pfrag, bv, acc[t]);
        }
        __syncthreads();   // #4: buf[nxt] safe to overwrite, P/S reusable
    }

    // ---- epilogue: 1/l normalize, gamma-gated residual, store f32 ----------
    const float g    = gamma[h];
    const float ginv = 1.0f / (g + 1.0f);
#pragma unroll
    for (int j = 0; j < 8; ++j) {
        const int M  = j + lhi * 8;
        const float l = lrow[sq][M];
        const float linv = (l > 0.0f) ? (1.0f / l) : 0.0f;
        const int lq = qrow0 + M;
#pragma unroll
        for (int t = 0; t < 8; ++t) {
            const int n = dwave + t * 16 + rlo;
            const float val = acc[t][j] * linv;
            const float xp  = XP[((size_t)b * LL + lq) * DD + n];
            out[(((size_t)b * HH + h) * LL + lq) * DD + n] = (g * val + xp) * ginv;
        }
    }
}

// ---------------------------------------------------------------------------
// Host launcher
// ---------------------------------------------------------------------------
extern "C" void kernel_launch(void* const* d_in, const int* in_sizes, int n_in,
                              void* d_out, int out_size, void* d_ws, size_t ws_size,
                              hipStream_t stream) {
    (void)in_sizes; (void)n_in; (void)out_size; (void)ws_size;

    const float* x      = (const float*)d_in[0];
    const float* y      = (const float*)d_in[1];
    /* x_mask (all true per setup) ignored: d_in[2] */
    const int*   y_mask = (const int*)d_in[3];
    const float* x_pos  = (const float*)d_in[4];
    const float* y_pos  = (const float*)d_in[5];
    const float* Wq     = (const float*)d_in[6];
    const float* bq     = (const float*)d_in[7];
    const float* Wk     = (const float*)d_in[8];
    const float* bk     = (const float*)d_in[9];
    const float* Wv     = (const float*)d_in[10];
    const float* bv     = (const float*)d_in[11];
    const float* Wp     = (const float*)d_in[12];
    const float* gamma  = (const float*)d_in[13];
    float* out = (float*)d_out;

    // ------- workspace layout -------
    char* ws = (char*)d_ws;
    const size_t QKV_ELEMS  = (size_t)HH * BB * LL * DD;   // 33,554,432
    const size_t ACT_ELEMS  = (size_t)BB * LL * DD;        //  4,194,304
    const size_t WQKV_ELEMS = (size_t)HH * DD * DD;        //  2,097,152
    const size_t WP_ELEMS   = (size_t)DD * DD;             //    262,144

    size_t off = 0;
    bf16_t* Qbf  = (bf16_t*)(ws + off); off += QKV_ELEMS * 2;
    bf16_t* Kbf  = (bf16_t*)(ws + off); off += QKV_ELEMS * 2;
    bf16_t* Vt   = (bf16_t*)(ws + off); off += QKV_ELEMS * 2;
    float*  XP   = (float*) (ws + off); off += ACT_ELEMS * 4;
    bf16_t* xaug = (bf16_t*)(ws + off); off += ACT_ELEMS * 2;
    bf16_t* yaug = (bf16_t*)(ws + off); off += ACT_ELEMS * 2;
    bf16_t* ybf  = (bf16_t*)(ws + off); off += ACT_ELEMS * 2;
    bf16_t* xbf  = (bf16_t*)(ws + off); off += ACT_ELEMS * 2;
    bf16_t* Wqbf = (bf16_t*)(ws + off); off += WQKV_ELEMS * 2;
    bf16_t* Wkbf = (bf16_t*)(ws + off); off += WQKV_ELEMS * 2;
    bf16_t* Wvbf = (bf16_t*)(ws + off); off += WQKV_ELEMS * 2;
    bf16_t* Wpbf = (bf16_t*)(ws + off); off += WP_ELEMS * 2;

    // ------- phase 1: conversions -------
    cvt_add_kernel<<<1024, 256, 0, stream>>>(x, x_pos, xaug, (int)ACT_ELEMS);
    cvt_add_kernel<<<1024, 256, 0, stream>>>(y, y_pos, yaug, (int)ACT_ELEMS);
    cvt_kernel<<<1024, 256, 0, stream>>>(y, ybf, (int)ACT_ELEMS);
    cvt_kernel<<<1024, 256, 0, stream>>>(x, xbf, (int)ACT_ELEMS);
    cvt_kernel<<<1024, 256, 0, stream>>>(Wq, Wqbf, (int)WQKV_ELEMS);
    cvt_kernel<<<1024, 256, 0, stream>>>(Wk, Wkbf, (int)WQKV_ELEMS);
    cvt_kernel<<<1024, 256, 0, stream>>>(Wv, Wvbf, (int)WQKV_ELEMS);
    cvt_kernel<<<512, 256, 0, stream>>>(Wp, Wpbf, (int)WP_ELEMS);

    // ------- phase 2: projections -------
    const int M = BB * LL;                  // 8192
    const int Nqkv = HH * DD;               // 4096
    const int blocksQKV = (M / 16) * (Nqkv / 512);   // 4096
    const int blocksP   = (M / 16) * (DD / 512);     // 512
    proj_kernel<<<blocksQKV, 256, 0, stream>>>(xaug, Wqbf, bq, (void*)Qbf, Nqkv, 0);
    proj_kernel<<<blocksQKV, 256, 0, stream>>>(yaug, Wkbf, bk, (void*)Kbf, Nqkv, 1);
    proj_kernel<<<blocksQKV, 256, 0, stream>>>(ybf,  Wvbf, bv, (void*)Vt,  Nqkv, 2);
    proj_kernel<<<blocksP,   256, 0, stream>>>(xbf,  Wpbf, nullptr, (void*)XP, DD, 3);

    // ------- phase 3: attention -------
    const int attnBlocks = HH * BB * (LL / QT);      // 2048
    attn_kernel<<<attnBlocks, 256, 0, stream>>>(Qbf, Kbf, Vt, XP, y_mask, gamma, out);
}
